// SupConLoss_73177652789547
// MI455X (gfx1250) — matrix operand order repaired and checked by reading, not compiled, not run on hardware
//
#include <hip/hip_runtime.h>
#include <hip/hip_bf16.h>
#include <math.h>

// ---------------------------------------------------------------------------
// SupConLoss, fused on CDNA5 (gfx1250, wave32, WMMA bf16 split-f32 GEMM).
//
// ws layout (needs 2*N*D*2 + 3*N*4 bytes = 16.1 MB for N=8192, D=512):
//   Phi  : N*D bf16   (high half of normalized projections)
//   Plo  : N*D bf16   (low  half: residual)
//   Ssum : N  f32     (sum_{j!=i} exp(sim - 1/T))
//   Tsum : N  f32     (sum_{pos} (sim - 1/T))
//   Csum : N  f32     (positive count)
// ---------------------------------------------------------------------------

typedef __attribute__((ext_vector_type(16))) __bf16 v16bf;
typedef __attribute__((ext_vector_type(8)))  float  v8f;
typedef __attribute__((ext_vector_type(4)))  unsigned int uv4;

#define INV_TEMP 14.285714285714285714f   // 1/0.07
#define LOG_EPS  1e-8f
#define NORM_EPS 1e-12f

#define JCHUNK 512   // columns of the Gram matrix handled per block
#define JGROUP 4     // 16-wide j-tiles register-blocked per k sweep

// ---------------------------------------------------------------------------
// Phase 1: L2-normalize each row, split into bf16 hi + bf16 lo residual.
// One wave (32 lanes) per row; strided, coalesced.
// ---------------------------------------------------------------------------
__global__ __launch_bounds__(32)
void scl_normalize_split(const float* __restrict__ X,
                         __bf16* __restrict__ Phi,
                         __bf16* __restrict__ Plo,
                         int D)
{
    const int row  = blockIdx.x;
    const int lane = threadIdx.x;
    const size_t base = (size_t)row * D;

    float ss = 0.0f;
    for (int j = lane; j < D; j += 32) {
        float x = X[base + j];
        ss += x * x;
    }
#pragma unroll
    for (int off = 16; off >= 1; off >>= 1)
        ss += __shfl_xor(ss, off, 32);

    const float scale = 1.0f / fmaxf(sqrtf(ss), NORM_EPS);

    for (int j = lane; j < D; j += 32) {
        float xs = X[base + j] * scale;
        __bf16 h  = (__bf16)xs;
        __bf16 lo = (__bf16)(xs - (float)h);
        Phi[base + j] = h;
        Plo[base + j] = lo;
    }
}

// ---------------------------------------------------------------------------
// Phase 2: zero the per-row accumulators.
// ---------------------------------------------------------------------------
__global__ void scl_zero(float* __restrict__ p, int n)
{
    int i = blockIdx.x * blockDim.x + threadIdx.x;
    if (i < n) p[i] = 0.0f;
}

// ---------------------------------------------------------------------------
// Phase 3: fused Gram-tile GEMM (bf16-split, f32 acc) + masked softmax stats.
// One wave per block. Block (bx, by): rows [by*16, by*16+16), cols
// [bx*JCHUNK, ..+JCHUNK). EXEC is all-ones everywhere around the WMMAs.
// ---------------------------------------------------------------------------
static __device__ inline v8f wmma_bf16(v16bf a, v16bf b, v8f c)
{
    return __builtin_amdgcn_wmma_f32_16x16x32_bf16(
        /*neg_a=*/false, a, /*neg_b=*/false, b,
        /*c_mod=*/(short)0, c, /*reuse_a=*/false, /*reuse_b=*/false);
}

__global__ __launch_bounds__(32)
void scl_sim_loss(const __bf16* __restrict__ Phi,
                  const __bf16* __restrict__ Plo,
                  const int*    __restrict__ tgt,
                  float* __restrict__ Ssum,
                  float* __restrict__ Tsum,
                  float* __restrict__ Csum,
                  int N, int D)
{
    const int lane = threadIdx.x;        // 0..31
    const int h    = lane >> 4;          // 0 or 1 (half-wave)
    const int l15  = lane & 15;
    const int ibase  = blockIdx.y * 16;
    const int jchunk = blockIdx.x * JCHUNK;

    // targets for the 8 rows this lane's C registers touch (row = r + 8*h)
    int tgtR[8];
#pragma unroll
    for (int r = 0; r < 8; ++r)
        tgtR[r] = tgt[ibase + r + 8 * h];

    // A operand row for this lane (A lane = M; this code gives half h the contiguous K 16h..16h+15)
    const int    arow  = ibase + l15;
    const size_t abase = (size_t)arow * D;

    float sAcc[8], tAcc[8], cAcc[8];
#pragma unroll
    for (int r = 0; r < 8; ++r) { sAcc[r] = 0.0f; tAcc[r] = 0.0f; cAcc[r] = 0.0f; }

    for (int jg = 0; jg < JCHUNK; jg += JGROUP * 16) {
        v8f c[JGROUP] = {};

        for (int k0 = 0; k0 < D; k0 += 32) {
            // --- A tile: elems 0..15 <- K = k0 + h*16 .. +15 (contiguous)
            union { v16bf v; uv4 q[2]; } ah, al;
            const size_t ak = abase + k0 + h * 16;
            ah.q[0] = *(const uv4*)(Phi + ak);
            ah.q[1] = *(const uv4*)(Phi + ak + 8);
            al.q[0] = *(const uv4*)(Plo + ak);
            al.q[1] = *(const uv4*)(Plo + ak + 8);

#pragma unroll
            for (int t = 0; t < JGROUP; ++t) {
                // --- B tile: column = row (jcol) of p; lane half selects K half
                const int    jcol = jchunk + jg + t * 16 + l15;
                const size_t bk   = (size_t)jcol * D + k0 + h * 16;
                v16bf bh = *(const v16bf*)(Phi + bk);   // 32B contiguous
                v16bf bl = *(const v16bf*)(Plo + bk);

                c[t] = wmma_bf16(ah.v, bh, c[t]);  // hi*hi
                c[t] = wmma_bf16(ah.v, bl, c[t]);  // hi*lo
                c[t] = wmma_bf16(al.v, bh, c[t]);  // lo*hi
            }
        }

        // --- epilogue: masked exp / positive sums, in registers
#pragma unroll
        for (int t = 0; t < JGROUP; ++t) {
            const int jcol = jchunk + jg + t * 16 + l15;
            const int tc   = tgt[jcol];
#pragma unroll
            for (int r = 0; r < 8; ++r) {
                const int   row  = ibase + r + 8 * h;
                const float e    = (c[t][r] - 1.0f) * INV_TEMP;  // sim - 1/T
                const bool  self = (jcol == row);
                sAcc[r] += self ? 0.0f : __expf(e);
                const bool pos = (!self) && (tc == tgtR[r]);
                tAcc[r] += pos ? e : 0.0f;
                cAcc[r] += pos ? 1.0f : 0.0f;
            }
        }
    }

    // reduce across the 16 lanes that share each row (xor stays within half)
#pragma unroll
    for (int r = 0; r < 8; ++r) {
#pragma unroll
        for (int off = 1; off < 16; off <<= 1) {
            sAcc[r] += __shfl_xor(sAcc[r], off, 32);
            tAcc[r] += __shfl_xor(tAcc[r], off, 32);
            cAcc[r] += __shfl_xor(cAcc[r], off, 32);
        }
    }

    if (l15 == 0) {   // lane 0 -> rows 0..7, lane 16 -> rows 8..15
#pragma unroll
        for (int r = 0; r < 8; ++r) {
            const int row = ibase + r + 8 * h;
            atomicAdd(&Ssum[row], sAcc[r]);
            atomicAdd(&Tsum[row], tAcc[r]);
            atomicAdd(&Csum[row], cAcc[r]);
        }
    }
}

// ---------------------------------------------------------------------------
// Phase 4: loss_i = T_i/cnt_i - log(S_i + eps), 0 where cnt == 0.
// ---------------------------------------------------------------------------
__global__ void scl_finalize(const float* __restrict__ Ssum,
                             const float* __restrict__ Tsum,
                             const float* __restrict__ Csum,
                             float* __restrict__ out, int N)
{
    int i = blockIdx.x * blockDim.x + threadIdx.x;
    if (i < N) {
        float cnt = Csum[i];
        out[i] = (cnt > 0.0f) ? (Tsum[i] / cnt - logf(Ssum[i] + LOG_EPS))
                              : 0.0f;
    }
}

// ---------------------------------------------------------------------------
extern "C" void kernel_launch(void* const* d_in, const int* in_sizes, int n_in,
                              void* d_out, int out_size, void* d_ws, size_t ws_size,
                              hipStream_t stream)
{
    const float* X   = (const float*)d_in[0];   // projections [N, D] f32
    const int*   tgt = (const int*)d_in[1];     // targets [N]
    float*       out = (float*)d_out;           // [N] f32

    const int N = in_sizes[1];
    const int D = in_sizes[0] / N;              // 8192, 512

    __bf16* Phi  = (__bf16*)d_ws;
    __bf16* Plo  = Phi + (size_t)N * D;
    float*  Ssum = (float*)(Plo + (size_t)N * D);
    float*  Tsum = Ssum + N;
    float*  Csum = Tsum + N;

    scl_zero<<<(3 * N + 255) / 256, 256, 0, stream>>>(Ssum, 3 * N);
    scl_normalize_split<<<N, 32, 0, stream>>>(X, Phi, Plo, D);

    dim3 grid(N / JCHUNK, N / 16);              // (16, 512) single-wave blocks
    scl_sim_loss<<<grid, 32, 0, stream>>>(Phi, Plo, tgt, Ssum, Tsum, Csum, N, D);

    scl_finalize<<<(N + 255) / 256, 256, 0, stream>>>(Ssum, Tsum, Csum, out, N);
}
